// ThreeLayerDirectedGCN_63986422776433
// MI455X (gfx1250) — compile-verified, matrix operand order checked by reading, not run on hardware
//
#include <hip/hip_runtime.h>
#include <hip/hip_bf16.h>
#include <stdint.h>

// ---------------------------------------------------------------------------
// Types for CDNA5 WMMA (wave32): v_wmma_f32_16x16x32_bf16
// ---------------------------------------------------------------------------
typedef __attribute__((ext_vector_type(16))) __bf16         v16bf;
typedef __attribute__((ext_vector_type(8)))  float          v8f;
typedef __attribute__((ext_vector_type(8)))  unsigned short ushort8;
typedef __attribute__((ext_vector_type(16))) unsigned short ushort16;

union BFrag {
    ushort8        u8[2];
    ushort16       u16;
    unsigned short us[16];
    v16bf          v;
};

__device__ __forceinline__ unsigned short f32_to_bf16_rne(float f) {
    unsigned int u = __float_as_uint(f);
    u += 0x7FFFu + ((u >> 16) & 1u);   // round-to-nearest-even
    return (unsigned short)(u >> 16);
}

// ---------------------------------------------------------------------------
// Small elementwise kernels
// ---------------------------------------------------------------------------
__global__ void zero_f32_kernel(float* __restrict__ p, int n) {
    int i = blockIdx.x * blockDim.x + threadIdx.x;
    if (i < n) p[i] = 0.0f;
}

__global__ void cvt_bf16_kernel(const float* __restrict__ src,
                                unsigned short* __restrict__ dst, int n) {
    int i = blockIdx.x * blockDim.x + threadIdx.x;
    if (i < n) dst[i] = f32_to_bf16_rne(src[i]);
}

// Fused ReLU + fp32->bf16 (next layer only consumes the bf16 copy)
__global__ void relu_cvt_kernel(const float* __restrict__ in,
                                unsigned short* __restrict__ out, int n) {
    int i = blockIdx.x * blockDim.x + threadIdx.x;
    if (i < n) out[i] = f32_to_bf16_rne(fmaxf(in[i], 0.0f));
}

__global__ void degree_kernel(const int* __restrict__ row, const int* __restrict__ col,
                              float* __restrict__ dr, float* __restrict__ dc, int E) {
    int e = blockIdx.x * blockDim.x + threadIdx.x;
    if (e < E) {
        atomicAdd(&dr[row[e]], 1.0f);
        atomicAdd(&dc[col[e]], 1.0f);
    }
}

__global__ void rsqrt_kernel(float* __restrict__ p, int n) {
    int i = blockIdx.x * blockDim.x + threadIdx.x;
    if (i < n) {
        float v = p[i];
        p[i] = (v > 0.0f) ? rsqrtf(v) : 0.0f;  // zero-degree nodes never gathered
    }
}

__global__ void sigmoid_kernel(const float* __restrict__ in, float* __restrict__ out, int n) {
    int i = blockIdx.x * blockDim.x + threadIdx.x;
    if (i < n) out[i] = 1.0f / (1.0f + expf(-in[i]));
}

// ---------------------------------------------------------------------------
// Pack weight matrix B[K x M] (f32, row-major) into WMMA-fragment order bf16:
//   Bp[kb][c][h][16] :
//     elements 0..7  = B[kb*32 + h*8 + j      ][c]   (j = 0..7)
//     elements 8..15 = B[kb*32 + h*8 + 16 + j ][c]   (j = 0..7)
// so a lane (half = h, column = c) loads its whole 16-bf16 fragment for
// k-block kb as one contiguous 32-byte vector (global_load_b128 pair).
// M must be a power of two (256 / 128 here); mlog2 = log2(M).
// ---------------------------------------------------------------------------
__global__ void pack_b_kernel(const float* __restrict__ B,
                              unsigned short* __restrict__ Bp,
                              int M, int mlog2, int total) {
    int i = blockIdx.x * blockDim.x + threadIdx.x;
    if (i >= total) return;
    const int idx = i & 15;
    const int h   = (i >> 4) & 1;
    const int c   = (i >> 5) & (M - 1);
    const int kb  = i >> (5 + mlog2);
    const int k   = kb * 32 + h * 8 + (idx & 7) + ((idx >> 3) << 4);
    Bp[i] = f32_to_bf16_rne(B[(size_t)k * M + c]);
}

// ---------------------------------------------------------------------------
// GEMM: C[N x M] = bf16(A[N x K]) @ bf16(B[K x M]) (+ bias), f32 accumulate.
// Block = 256 threads = 8 waves; block tile 128 rows x 64 cols.
// Each wave: 16 rows x 64 cols = 4 accumulators; A fragment reused 4x.
// No LDS: A fragments are two 16B contiguous global loads, B fragments are
// one 32B contiguous global load from the pre-packed buffer (L2-resident).
// A-fragment layout (ISA 7.12.2, 16-bit A 16x32):
//   lanes 0-15:  row = lane,     K = k0+0..7  and k0+16..23
//   lanes 16-31: row = lane-16,  K = k0+8..15 and k0+24..31
// C layout: VGPR v -> row v (lanes 0-15) / row v+8 (lanes 16-31), col = lane&15.
// ---------------------------------------------------------------------------
__launch_bounds__(256, 4)
__global__ void gemm_bf16_wmma_kernel(const unsigned short* __restrict__ A,
                                      const unsigned short* __restrict__ Bp,
                                      const float* __restrict__ bias,
                                      float* __restrict__ C,
                                      int Nrows, int K, int M)
{
    const int c0   = blockIdx.x * 64;
    const int wave = threadIdx.x >> 5;
    const int lane = threadIdx.x & 31;
    const int r0   = blockIdx.y * 128 + wave * 16;
    if (r0 >= Nrows) return;                 // whole-wave skip (EXEC all-1 for WMMA)

    const int half = lane >> 4;              // 0: K 0..7/16..23   1: K 8..15/24..31
    const int l15  = lane & 15;

    v8f acc0 = {}, acc1 = {}, acc2 = {}, acc3 = {};

    const unsigned short* arow  = A + (size_t)(r0 + l15) * K;
    // fragment offset: ((kb*M + c)*2 + half)*16 = kb*M*32 + c*32 + half*16
    const unsigned short* bbase = Bp + (size_t)(c0 + l15) * 32 + half * 16;
    const size_t bKB = (size_t)M * 32;       // elements per k-block

    const int nkb = K >> 5;
    for (int kb = 0; kb < nkb; ++kb) {
        const int ka = kb * 32 + half * 8;

        BFrag a;
        a.u8[0] = *(const ushort8*)(arow + ka);        // global_load_b128
        a.u8[1] = *(const ushort8*)(arow + ka + 16);   // global_load_b128

        const unsigned short* bp = bbase + (size_t)kb * bKB;
        BFrag b0, b1, b2, b3;
        b0.u16 = *(const ushort16*)(bp);               // 32B contiguous
        b1.u16 = *(const ushort16*)(bp + 512);         // +16 columns
        b2.u16 = *(const ushort16*)(bp + 1024);
        b3.u16 = *(const ushort16*)(bp + 1536);

        acc0 = __builtin_amdgcn_wmma_f32_16x16x32_bf16(false, a.v, false, b0.v,
                                                       (short)0, acc0, false, false);
        acc1 = __builtin_amdgcn_wmma_f32_16x16x32_bf16(false, a.v, false, b1.v,
                                                       (short)0, acc1, false, false);
        acc2 = __builtin_amdgcn_wmma_f32_16x16x32_bf16(false, a.v, false, b2.v,
                                                       (short)0, acc2, false, false);
        acc3 = __builtin_amdgcn_wmma_f32_16x16x32_bf16(false, a.v, false, b3.v,
                                                       (short)0, acc3, false, false);
    }

    const int rbase = r0 + half * 8;
    v8f accs[4] = {acc0, acc1, acc2, acc3};
#pragma unroll
    for (int t = 0; t < 4; ++t) {
        const int c    = c0 + t * 16 + l15;
        const float bv = bias ? bias[c] : 0.0f;
#pragma unroll
        for (int v = 0; v < 8; ++v) {
            const int rr = rbase + v;
            if (rr < Nrows) C[(size_t)rr * M + c] = accs[t][v] + bv;
        }
    }
}

// ---------------------------------------------------------------------------
// Edge scatter: agg[col] += norm(row,col) * xw[row], vectorized float4 gather,
// f32 global atomics for the scatter-add. 2^shift = D/4 threads per edge.
// ---------------------------------------------------------------------------
__global__ void scatter_edges_kernel(const int* __restrict__ row, const int* __restrict__ col,
                                     const float* __restrict__ dri, const float* __restrict__ dci,
                                     const float* __restrict__ xw, float* __restrict__ agg,
                                     int E, int D, int shift)
{
    const int gid = blockIdx.x * blockDim.x + threadIdx.x;
    const int e   = gid >> shift;
    if (e >= E) return;
    const int f = (gid - (e << shift)) << 2;

    const int r = row[e];
    const int c = col[e];
    const float nrm = dri[r] * dci[c];

    const float4 v = *(const float4*)(xw + (size_t)r * D + f);
    float* dst = agg + (size_t)c * D + f;
    atomicAdd(dst + 0, nrm * v.x);
    atomicAdd(dst + 1, nrm * v.y);
    atomicAdd(dst + 2, nrm * v.z);
    atomicAdd(dst + 3, nrm * v.w);
}

// ---------------------------------------------------------------------------
// Host orchestration
// ---------------------------------------------------------------------------
extern "C" void kernel_launch(void* const* d_in, const int* in_sizes, int n_in,
                              void* d_out, int out_size, void* d_ws, size_t ws_size,
                              hipStream_t stream)
{
    (void)n_in; (void)ws_size; (void)out_size;

    const float* x   = (const float*)d_in[0];
    const int*   ei  = (const int*)d_in[1];     // [2, E] int32
    const float* w1  = (const float*)d_in[3];
    const float* lw1 = (const float*)d_in[4];
    const float* lb1 = (const float*)d_in[5];
    const float* w2  = (const float*)d_in[6];
    const float* lw2 = (const float*)d_in[7];
    const float* lb2 = (const float*)d_in[8];
    const float* w3  = (const float*)d_in[9];
    const float* lw3 = (const float*)d_in[10];
    const float* lb3 = (const float*)d_in[11];

    const int D_IN = 256, D_HID = 256, D_OUT = 128;
    const int N = in_sizes[0] / D_IN;
    const int E = in_sizes[1] / 2;
    const int* row = ei;
    const int* col = ei + E;

    // Workspace carving (256B aligned slots)
    char*  ws  = (char*)d_ws;
    size_t off = 0;
    auto carve = [&](size_t bytes) -> char* {
        char* p = ws + off;
        off = (off + bytes + 255) & ~(size_t)255;
        return p;
    };
    float* degr = (float*)carve((size_t)N * 4);
    float* degc = (float*)carve((size_t)N * 4);
    float* xw   = (float*)carve((size_t)N * D_HID * 4);
    float* agg  = (float*)carve((size_t)N * D_HID * 4);
    unsigned short* abf  = (unsigned short*)carve((size_t)N * D_HID * 2);
    unsigned short* w1p  = (unsigned short*)carve((size_t)D_IN  * D_HID * 2);
    unsigned short* lw1p = (unsigned short*)carve((size_t)D_IN  * D_HID * 2);
    unsigned short* w2p  = (unsigned short*)carve((size_t)D_HID * D_HID * 2);
    unsigned short* lw2p = (unsigned short*)carve((size_t)D_HID * D_HID * 2);
    unsigned short* w3p  = (unsigned short*)carve((size_t)D_HID * D_OUT * 2);
    unsigned short* lw3p = (unsigned short*)carve((size_t)D_HID * D_OUT * 2);

    auto cdiv = [](int a, int b) { return (a + b - 1) / b; };

    // --- degrees and symmetric norm ---
    zero_f32_kernel<<<cdiv(N, 256), 256, 0, stream>>>(degr, N);
    zero_f32_kernel<<<cdiv(N, 256), 256, 0, stream>>>(degc, N);
    degree_kernel<<<cdiv(E, 256), 256, 0, stream>>>(row, col, degr, degc, E);
    rsqrt_kernel<<<cdiv(N, 256), 256, 0, stream>>>(degr, N);
    rsqrt_kernel<<<cdiv(N, 256), 256, 0, stream>>>(degc, N);

    // --- weights: fp32 -> bf16 + WMMA-fragment packing (tiny, L2-resident) ---
    pack_b_kernel<<<cdiv(D_IN * D_HID, 256), 256, 0, stream>>>(w1,  w1p,  D_HID, 8, D_IN * D_HID);
    pack_b_kernel<<<cdiv(D_IN * D_HID, 256), 256, 0, stream>>>(lw1, lw1p, D_HID, 8, D_IN * D_HID);
    pack_b_kernel<<<cdiv(D_HID * D_HID, 256), 256, 0, stream>>>(w2,  w2p,  D_HID, 8, D_HID * D_HID);
    pack_b_kernel<<<cdiv(D_HID * D_HID, 256), 256, 0, stream>>>(lw2, lw2p, D_HID, 8, D_HID * D_HID);
    pack_b_kernel<<<cdiv(D_HID * D_OUT, 256), 256, 0, stream>>>(w3,  w3p,  D_OUT, 7, D_HID * D_OUT);
    pack_b_kernel<<<cdiv(D_HID * D_OUT, 256), 256, 0, stream>>>(lw3, lw3p, D_OUT, 7, D_HID * D_OUT);

    const dim3 gHID(D_HID / 64, cdiv(N, 128));   // M=256 GEMMs
    const dim3 gOUT(D_OUT / 64, cdiv(N, 128));   // M=128 GEMM

    // ---------------- Layer 1 (input = x, D_IN -> D_HID, ReLU) ----------------
    cvt_bf16_kernel<<<cdiv(N * D_IN, 256), 256, 0, stream>>>(x, abf, N * D_IN);
    gemm_bf16_wmma_kernel<<<gHID, 256, 0, stream>>>(abf, w1p,  nullptr, xw,  N, D_IN, D_HID);
    gemm_bf16_wmma_kernel<<<gHID, 256, 0, stream>>>(abf, lw1p, lb1,     agg, N, D_IN, D_HID);
    scatter_edges_kernel<<<cdiv(E * 64, 256), 256, 0, stream>>>(row, col, degr, degc,
                                                                xw, agg, E, D_HID, 6);
    relu_cvt_kernel<<<cdiv(N * D_HID, 256), 256, 0, stream>>>(agg, abf, N * D_HID);

    // ---------------- Layer 2 (D_HID -> D_HID, ReLU) ----------------
    gemm_bf16_wmma_kernel<<<gHID, 256, 0, stream>>>(abf, w2p,  nullptr, xw,  N, D_HID, D_HID);
    gemm_bf16_wmma_kernel<<<gHID, 256, 0, stream>>>(abf, lw2p, lb2,     agg, N, D_HID, D_HID);
    scatter_edges_kernel<<<cdiv(E * 64, 256), 256, 0, stream>>>(row, col, degr, degc,
                                                                xw, agg, E, D_HID, 6);
    relu_cvt_kernel<<<cdiv(N * D_HID, 256), 256, 0, stream>>>(agg, abf, N * D_HID);

    // ---------------- Layer 3 (D_HID -> D_OUT, Sigmoid -> d_out) ----------------
    gemm_bf16_wmma_kernel<<<gOUT, 256, 0, stream>>>(abf, w3p,  nullptr, xw,  N, D_HID, D_OUT);
    gemm_bf16_wmma_kernel<<<gOUT, 256, 0, stream>>>(abf, lw3p, lb3,     agg, N, D_HID, D_OUT);
    scatter_edges_kernel<<<cdiv(E * 32, 256), 256, 0, stream>>>(row, col, degr, degc,
                                                                xw, agg, E, D_OUT, 5);
    sigmoid_kernel<<<cdiv(N * D_OUT, 256), 256, 0, stream>>>(agg, (float*)d_out, N * D_OUT);
}
